// EarthAttention2D_81999515615773
// MI455X (gfx1250) — matrix operand order repaired and checked by reading, not compiled, not run on hardware
//
#include <hip/hip_runtime.h>

// ---------------------------------------------------------------------------
// EarthAttention2D for MI455X (gfx1250, wave32, WMMA f16->f32 16x16x32)
// One block per attention window; fully fused in LDS (~166KB of 320KB/WGP).
// x tile staged into LDS with global_load_async_to_lds_b128 (ASYNCcnt path).
// ---------------------------------------------------------------------------

typedef _Float16 h8  __attribute__((ext_vector_type(8)));
typedef _Float16 h16 __attribute__((ext_vector_type(16)));
typedef float    f8  __attribute__((ext_vector_type(8)));

constexpr int DIM   = 192;
constexpr int HEADS = 6;
constexpr int HD    = 32;
constexpr int NW    = 72;    // tokens per window
constexpr int NPAD  = 80;    // M padded to 5x16
constexpr int KPAD  = 96;    // K padded to 3x32 for P@V
constexpr int TOW   = 31;
constexpr int LONW  = 30;

__device__ __forceinline__ h16 cat8(h8 lo, h8 hi) {
  return __builtin_shufflevector(lo, hi, 0,1,2,3,4,5,6,7,8,9,10,11,12,13,14,15);
}

// A-fragment (16x32 f16): lane<16 holds row m=lane, K = {0..7, 16..23};
// lane>=16 same row, K = {8..15, 24..31}. Two 16B chunks at kbase+d0, +16.
__device__ __forceinline__ h16 loadA(const _Float16* base, int row, int stride,
                                     int kbase, int hi16) {
  const _Float16* p = base + row * stride + kbase + (hi16 ? 8 : 0);
  h8 lo = *(const h8*)(p);
  h8 hi = *(const h8*)(p + 16);
  return cat8(lo, hi);
}

// B-fragment (32x16 f16): lane holds column n=lane%16; lane<16 -> K {0..7,16..23},
// lane>=16 -> K {8..15,24..31}. base must be N-major (stride in K units).
__device__ __forceinline__ h16 loadB(const _Float16* base, int col, int stride,
                                     int kbase, int hi16) {
  const _Float16* p = base + col * stride + kbase + (hi16 ? 8 : 0);
  h8 lo = *(const h8*)(p);
  h8 hi = *(const h8*)(p + 16);
  return cat8(lo, hi);
}

__device__ __forceinline__ f8 wmma16(h16 a, h16 b, f8 c) {
  return __builtin_amdgcn_wmma_f32_16x16x32_f16(false, a, false, b,
                                                (short)0, c, false, false);
}

// byte offset of an LDS pointer within the workgroup LDS allocation
__device__ __forceinline__ unsigned lds_byte_off(const void* p) {
  return (unsigned)(size_t)(__attribute__((address_space(3))) const char*)p;
}

// ---- prep: transpose + f16-convert weights (B-frags need N-major layout) ---
__global__ void prep_weights(const float* __restrict__ w_qkv,
                             const float* __restrict__ w_out,
                             _Float16* __restrict__ wqkvT,
                             _Float16* __restrict__ woutT) {
  int idx = blockIdx.x * blockDim.x + threadIdx.x;
  if (idx < 3 * DIM * DIM) {                    // (576,192) <- (192,576)
    int n = idx / DIM, k = idx % DIM;
    wqkvT[idx] = (_Float16)w_qkv[k * (3 * DIM) + n];
  } else if (idx < 3 * DIM * DIM + DIM * DIM) { // (192,192) transpose
    int r = idx - 3 * DIM * DIM;
    int n = r / DIM, k = r % DIM;
    woutT[r] = (_Float16)w_out[k * DIM + n];
  }
}

__global__ __launch_bounds__(256) void earth_attn_kernel(
    const float* __restrict__ x,      const float* __restrict__ mask,
    const float* __restrict__ b_qkv,  const float* __restrict__ b_out,
    const float* __restrict__ table,
    const _Float16* __restrict__ wqkvT, const _Float16* __restrict__ woutT,
    float* __restrict__ out) {
  extern __shared__ char smem[];
  float*    sS = (float*)smem;                        // [NPAD*NPAD] scores (f32)
  _Float16* xs = (_Float16*)(smem + NPAD * NPAD * 4); // [NPAD*DIM] x-f16, later attn-out
  _Float16* qs = xs + NPAD * DIM;                     // [NPAD*DIM]
  _Float16* ks = qs + NPAD * DIM;                     // [NPAD*DIM]
  _Float16* vT = ks + NPAD * DIM;                     // [DIM*KPAD]  v transposed
  _Float16* pp = vT + DIM * KPAD;                     // [NPAD*KPAD] probs f16

  const int tid  = threadIdx.x;
  const int lane = tid & 31;
  const int wave = tid >> 5;          // 8 waves
  const int hi16 = (lane >= 16);
  const int ln16 = lane & 15;
  const int hi8  = hi16 ? 8 : 0;
  const int b_   = blockIdx.x;

  // ---------------- init ----------------------------------------------------
  // 1) async-copy x (f32, 55KB) into LDS staging. Staging aliases q|k region,
  //    which is dead until phase 1 writes it (after the convert + barrier).
  const float* xg = x + (size_t)b_ * NW * DIM;
  {
    const unsigned stage0 = lds_byte_off(qs);
    for (int i = tid; i < NW * DIM / 4; i += 256) {     // 16B per chunk
      unsigned lofs = stage0 + (unsigned)i * 16u;
      unsigned gofs = (unsigned)i * 16u;
      asm volatile("global_load_async_to_lds_b128 %0, %1, %2"
                   :: "v"(lofs), "v"(gofs), "s"(xg) : "memory");
    }
  }
  // 2) warm L2 with the f16 weights while the async copy is in flight
  if (b_ < 64) {
    for (int i = tid; i < 1728; i += 256)
      __builtin_prefetch((const char*)wqkvT + i * 128, 0, 3);
    for (int i = tid; i < 576; i += 256)
      __builtin_prefetch((const char*)woutT + i * 128, 0, 3);
  }
  // 3) zero pad regions (disjoint from staging)
  for (int i = tid; i < (NPAD - NW) * DIM; i += 256) xs[NW * DIM + i] = (_Float16)0.f;
  for (int i = tid; i < DIM * (KPAD - NPAD); i += 256) {   // vT K-tail rows
    int c = i / (KPAD - NPAD), m = NPAD + i % (KPAD - NPAD);
    vT[c * KPAD + m] = (_Float16)0.f;
  }
  for (int i = tid; i < (NPAD - NW) * KPAD; i += 256)      // pp pad rows
    pp[NW * KPAD + i] = (_Float16)0.f;

  asm volatile("s_wait_asynccnt 0x0" ::: "memory");
  __syncthreads();

  // 4) convert staged f32 -> f16 xs
  {
    const float* stage = (const float*)qs;
    for (int i = tid; i < NW * DIM; i += 256) xs[i] = (_Float16)stage[i];
  }
  __syncthreads();

  // ---------------- phase 1: fused QKV projection (72x192 @ 192x576) -------
  const float scale = 0.17677669529663687f;  // hd^-0.5
  for (int id = wave; id < 5 * 36; id += 8) {
    int mt = id / 36, nt = id % 36;
    f8 acc = {0.f,0.f,0.f,0.f,0.f,0.f,0.f,0.f};
#pragma unroll
    for (int kk = 0; kk < 6; ++kk) {
      h16 a = loadA(xs, mt * 16 + ln16, DIM, kk * 32, hi16);
      h16 b = loadB(wqkvT, nt * 16 + ln16, DIM, kk * 32, hi16);
      acc = wmma16(a, b, acc);
    }
    // section (q/k/v) is identical for all lanes of the wave -> scalar branch
    int sec  = __builtin_amdgcn_readfirstlane(nt / 12);
    int cloc = (nt % 12) * 16 + ln16;
    float bv = b_qkv[sec * DIM + cloc];
    if (sec == 0) {
#pragma unroll
      for (int r = 0; r < 8; ++r)
        qs[(mt * 16 + r + hi8) * DIM + cloc] = (_Float16)((acc[r] + bv) * scale);
    } else if (sec == 1) {
#pragma unroll
      for (int r = 0; r < 8; ++r)
        ks[(mt * 16 + r + hi8) * DIM + cloc] = (_Float16)(acc[r] + bv);
    } else {
#pragma unroll
      for (int r = 0; r < 8; ++r)
        vT[cloc * KPAD + (mt * 16 + r + hi8)] = (_Float16)(acc[r] + bv);
    }
  }
  __syncthreads();

  // ---------------- phase 2: per-head attention -----------------------------
  const int bd31 = b_ / TOW;          // reference's tiled-pos batch offset
  const int t31  = b_ % TOW;          // tow index into bias table
  const float* mrow = mask + (size_t)(b_ % LONW) * NW * NW;

  for (int h = 0; h < HEADS; ++h) {
    // S = q @ k^T   (5x5 tiles of 16x16, K = hd = 32)
    for (int id = wave; id < 25; id += 8) {
      int mt = id / 5, nt = id % 5;
      h16 a = loadA(qs, mt * 16 + ln16, DIM, h * HD, hi16);
      h16 b = loadB(ks, nt * 16 + ln16, DIM, h * HD, hi16);
      f8 acc = {0.f,0.f,0.f,0.f,0.f,0.f,0.f,0.f};
      acc = wmma16(a, b, acc);
#pragma unroll
      for (int r = 0; r < 8; ++r)
        sS[(mt * 16 + r + hi8) * NPAD + nt * 16 + ln16] = acc[r];
    }
    __syncthreads();

    // earth-position bias + mask + softmax: one row per wave-iteration,
    // 32 lanes cover the 72 columns (<=3 each), shuffle reductions.
    for (int i = wave; i < NW; i += 8) {
      int qb = i * NW;
      float ev[3];
      float mx = -1e30f;
#pragma unroll
      for (int c = 0; c < 3; ++c) {
        int j = lane + 32 * c;
        float v = -1e30f;
        if (j < NW) {
          int aidx = (qb + j) * 60 + bd31;
          aidx -= (aidx / 5184) * 5184;           // tile(pos, rep) wrap
          int i2 = aidx / NW, j2 = aidx - i2 * NW;
          int row = 23 * ((i2 / 12) + 6 * (j2 / 12)) + (i2 % 12) - (j2 % 12) + 11;
          v = sS[i * NPAD + j] + table[(row * TOW + t31) * HEADS + h] + mrow[qb + j];
        }
        ev[c] = v;
        mx = fmaxf(mx, v);
      }
#pragma unroll
      for (int off = 16; off > 0; off >>= 1) mx = fmaxf(mx, __shfl_xor(mx, off, 32));
      float sum = 0.f;
#pragma unroll
      for (int c = 0; c < 3; ++c) {
        int j = lane + 32 * c;
        float e = (j < NW) ? __expf(ev[c] - mx) : 0.f;
        ev[c] = e;
        sum += e;
      }
#pragma unroll
      for (int off = 16; off > 0; off >>= 1) sum += __shfl_xor(sum, off, 32);
      float inv = 1.f / sum;
#pragma unroll
      for (int c = 0; c < 3; ++c) {
        int j = lane + 32 * c;
        if (j < KPAD) pp[i * KPAD + j] = (_Float16)(ev[c] * inv); // pads get 0
      }
    }
    __syncthreads();

    // O_h = P @ V_h  (5x2 tiles, K = 96) -> attn-out (reuses xs buffer)
    for (int id = wave; id < 10; id += 8) {
      int mt = id / 2, nt = id % 2;
      f8 acc = {0.f,0.f,0.f,0.f,0.f,0.f,0.f,0.f};
#pragma unroll
      for (int kk = 0; kk < 3; ++kk) {
        h16 a = loadA(pp, mt * 16 + ln16, KPAD, kk * 32, hi16);
        h16 b = loadB(vT, h * HD + nt * 16 + ln16, KPAD, kk * 32, hi16);
        acc = wmma16(a, b, acc);
      }
#pragma unroll
      for (int r = 0; r < 8; ++r)
        xs[(mt * 16 + r + hi8) * DIM + h * HD + nt * 16 + ln16] = (_Float16)acc[r];
    }
    __syncthreads();
  }

  // ---------------- phase 3: output projection (72x192 @ 192x192) ----------
  for (int id = wave; id < 60; id += 8) {
    int mt = id / 12, nt = id % 12;
    f8 acc = {0.f,0.f,0.f,0.f,0.f,0.f,0.f,0.f};
#pragma unroll
    for (int kk = 0; kk < 6; ++kk) {
      h16 a = loadA(xs, mt * 16 + ln16, DIM, kk * 32, hi16);
      h16 b = loadB(woutT, nt * 16 + ln16, DIM, kk * 32, hi16);
      acc = wmma16(a, b, acc);
    }
    int ncol = nt * 16 + ln16;
    float bo = b_out[ncol];
#pragma unroll
    for (int r = 0; r < 8; ++r) {
      int m = mt * 16 + r + hi8;
      if (m < NW)
        out[((size_t)b_ * NW + m) * DIM + ncol] = acc[r] + bo;
    }
  }
}

extern "C" void kernel_launch(void* const* d_in, const int* in_sizes, int n_in,
                              void* d_out, int out_size, void* d_ws, size_t ws_size,
                              hipStream_t stream) {
  const float* x      = (const float*)d_in[0];
  const float* mask   = (const float*)d_in[1];
  const float* w_qkv  = (const float*)d_in[2];
  const float* b_qkv  = (const float*)d_in[3];
  const float* w_out  = (const float*)d_in[4];
  const float* b_out  = (const float*)d_in[5];
  const float* table  = (const float*)d_in[6];
  float* out = (float*)d_out;

  _Float16* wqkvT = (_Float16*)d_ws;              // 576*192 f16
  _Float16* woutT = wqkvT + 3 * DIM * DIM;        // 192*192 f16

  int B_ = in_sizes[0] / (NW * DIM);              // 1860 windows

  int prep_total = 3 * DIM * DIM + DIM * DIM;
  prep_weights<<<(prep_total + 255) / 256, 256, 0, stream>>>(w_qkv, w_out, wqkvT, woutT);

  size_t smem = (size_t)NPAD * NPAD * sizeof(float) +
                sizeof(_Float16) * (3 * NPAD * DIM + DIM * KPAD + NPAD * KPAD);
  (void)hipFuncSetAttribute((const void*)earth_attn_kernel,
                            hipFuncAttributeMaxDynamicSharedMemorySize, (int)smem);
  earth_attn_kernel<<<B_, 256, smem, stream>>>(x, mask, b_qkv, b_out, table,
                                               wqkvT, woutT, out);
}
